// Regressor_25125558682089
// MI455X (gfx1250) — matrix operand order repaired and checked by reading, not compiled, not run on hardware
//
#include <hip/hip_runtime.h>
#include <hip/hip_bf16.h>

// ---------------------------------------------------------------------------
// CDNA5 (gfx1250) implementation.
// GEMMs: operands pre-packed to bf16 with K padded to a multiple of 32
// (zero-filled) and W pre-transposed, so the WMMA K-loop is branch-free:
// 6 aligned b128 loads -> 2x v_wmma_f32_16x16x32_bf16 per iteration.
// Fragment layouts per CDNA5 ISA §7.12.2:
//   A 16x32 : lane(idx,half) elem e -> k = (e&7) + (e>=8?16:0) + 8*half
//   B 32x16 : lane(idx,half) elem e -> k = (e&7) + (e>=8?16:0) + 8*half
//   C 16x16 : lane(idx,half) VGPR r -> m = r + 8*half, n = idx
// ---------------------------------------------------------------------------

typedef __attribute__((ext_vector_type(16))) __bf16 v16bf;
typedef __attribute__((ext_vector_type(8)))  __bf16 v8bf;
typedef __attribute__((ext_vector_type(8)))  float  v8f;

#define EPS_BN 1e-5f

static __host__ __device__ inline int roundup32(int k) { return (k + 31) & ~31; }

// ---------------------------------------------------------------------------
// Packing kernels: fp32 -> bf16, K padded to ldK with zeros.
// ---------------------------------------------------------------------------
__global__ void pack_a_kernel(const float* __restrict__ A, __bf16* __restrict__ Ab,
                              int K, int ldK) {
    int r = blockIdx.x;
    int k = blockIdx.y * 256 + threadIdx.x;
    if (k >= ldK) return;
    Ab[(long)r * ldK + k] = (k < K) ? (__bf16)A[(long)r * K + k] : (__bf16)0.0f;
}

// WT[n][k] = W[k][n]  (transpose + convert, zero-padded to ldK)
__global__ void pack_wt_kernel(const float* __restrict__ W, __bf16* __restrict__ WT,
                               int K, int N, int ldK) {
    int n = blockIdx.x;
    int k = blockIdx.y * 256 + threadIdx.x;
    if (k >= ldK) return;
    WT[(long)n * ldK + k] = (k < K) ? (__bf16)W[(long)k * N + n] : (__bf16)0.0f;
}

// ---------------------------------------------------------------------------
// WMMA GEMM: D[M x N] = epilogue(Ab[M x ldK] @ WT[N x ldK]^T)
// One wave -> 32x16 output tile (2 WMMAs sharing the B fragment).
// 4 waves per block along N. Requires M%32==0, N%64==0, ldK%32==0.
// Output goes to fp32 (out_f32) or packed bf16 (out_bf16) — exactly one set.
// ---------------------------------------------------------------------------
__global__ void gemm_bf16_wmma(const __bf16* __restrict__ Ab,
                               const __bf16* __restrict__ WT,
                               const float* __restrict__ bias,
                               const float* __restrict__ bn_g,
                               const float* __restrict__ bn_b,
                               const float* __restrict__ bn_m,
                               const float* __restrict__ bn_v,
                               float* __restrict__ out_f32,
                               __bf16* __restrict__ out_bf16,
                               int M, int ldK, int N, int do_relu) {
    const int lane = threadIdx.x & 31;
    const int wave = threadIdx.x >> 5;
    const int idx  = lane & 15;
    const int half = lane >> 4;

    const int tn = blockIdx.x * 4 + wave;       // N tile (16 cols)
    const int tm = blockIdx.y;                  // M tile (32 rows)
    if (tn * 16 >= N) return;                   // wave-uniform

    const int col = tn * 16 + idx;

    const __bf16* __restrict__ ap0 = Ab + (long)(tm * 32 + idx) * ldK + half * 8;
    const __bf16* __restrict__ ap1 = ap0 + (long)16 * ldK;   // rows +16
    const __bf16* __restrict__ bp  = WT + (long)col * ldK + half * 8;

    v8f acc0 = {}, acc1 = {};
    for (int k0 = 0; k0 < ldK; k0 += 32) {
        if (k0 + 128 < ldK) {                   // uniform guard: prefetch ahead
            __builtin_prefetch(ap0 + k0 + 128, 0, 3);
            __builtin_prefetch(ap1 + k0 + 128, 0, 3);
            __builtin_prefetch(bp  + k0 + 128, 0, 3);
        }
        v8bf b0 = *(const v8bf*)(bp  + k0);      // e 0..7  : k = k0+e+8h
        v8bf b1 = *(const v8bf*)(bp  + k0 + 16); // e 8..15 : k = k0+16+(e-8)+8h
        v8bf x0 = *(const v8bf*)(ap0 + k0);      // e 0..7  : k = k0+e+8h
        v8bf x1 = *(const v8bf*)(ap0 + k0 + 16); // e 8..15 : k = k0+16+(e-8)+8h
        v8bf y0 = *(const v8bf*)(ap1 + k0);
        v8bf y1 = *(const v8bf*)(ap1 + k0 + 16);

        v16bf bfrag = __builtin_shufflevector(b0, b1, 0,1,2,3,4,5,6,7,8,9,10,11,12,13,14,15);
        v16bf afrag = __builtin_shufflevector(x0, x1, 0,1,2,3,4,5,6,7,8,9,10,11,12,13,14,15);
        v16bf cfrag = __builtin_shufflevector(y0, y1, 0,1,2,3,4,5,6,7,8,9,10,11,12,13,14,15);

        acc0 = __builtin_amdgcn_wmma_f32_16x16x32_bf16(
            false, afrag, false, bfrag, (short)0, acc0, false, false);
        acc1 = __builtin_amdgcn_wmma_f32_16x16x32_bf16(
            false, cfrag, false, bfrag, (short)0, acc1, false, false);
    }

    // Epilogue (per output column `col`)
    float bv = bias ? bias[col] : 0.0f;
    float g = 1.0f, bb = 0.0f, mm = 0.0f, inv = 1.0f;
    const bool bn = (bn_g != nullptr);
    if (bn) {
        g = bn_g[col]; bb = bn_b[col]; mm = bn_m[col];
        inv = rsqrtf(bn_v[col] + EPS_BN);
    }
#pragma unroll
    for (int r = 0; r < 8; ++r) {
        int m0 = tm * 32 + r + half * 8;
        int m1 = m0 + 16;
        float x = acc0[r] + bv;
        float y = acc1[r] + bv;
        if (bn) { x = (x - mm) * inv * g + bb; y = (y - mm) * inv * g + bb; }
        if (do_relu) { x = fmaxf(x, 0.0f); y = fmaxf(y, 0.0f); }
        if (out_f32) {
            out_f32[(long)m0 * N + col] = x;
            out_f32[(long)m1 * N + col] = y;
        } else {
            out_bf16[(long)m0 * N + col] = (__bf16)x;
            out_bf16[(long)m1 * N + col] = (__bf16)y;
        }
    }
}

// ---------------------------------------------------------------------------
// Utility / graph kernels
// ---------------------------------------------------------------------------
__global__ void fill0_kernel(float* __restrict__ p, long n) {
    long t = (long)blockIdx.x * blockDim.x + threadIdx.x;
    if (t < n) p[t] = 0.0f;
}

__global__ void deg_kernel(const int* __restrict__ dst, float* __restrict__ deg,
                           int n_edges) {
    int e = blockIdx.x * blockDim.x + threadIdx.x;
    if (e < n_edges) atomicAdd(&deg[dst[e]], 1.0f);
}

// s[dst[e]*F + f] += h[src[e]*F + f]   (F = 1<<logF)
__global__ void scatter_add_kernel(const float* __restrict__ h,
                                   const int* __restrict__ src,
                                   const int* __restrict__ dst,
                                   float* __restrict__ s,
                                   long n_edges, int logF) {
    long t = (long)blockIdx.x * blockDim.x + threadIdx.x;
    if (t >= (n_edges << logF)) return;
    long e = t >> logF;
    int  f = (int)(t & ((1 << logF) - 1));
    atomicAdd(&s[((long)dst[e] << logF) + f], h[((long)src[e] << logF) + f]);
}

// in-place: s[n][f] = deg[n]>0 ? s[n][f]/max(deg,1) : h[n][f]
__global__ void agg_kernel(const float* __restrict__ h,
                           const float* __restrict__ deg,
                           float* __restrict__ s,
                           long n_nodes, int logF) {
    long t = (long)blockIdx.x * blockDim.x + threadIdx.x;
    if (t >= (n_nodes << logF)) return;
    long n = t >> logF;
    float d = deg[n];
    s[t] = (d > 0.0f) ? s[t] / fmaxf(d, 1.0f) : h[t];
}

// GCN layer 1 (K == 1): h1[n][f] = relu(agg1[n] * Wg1[f] + bg1[f])
__global__ void dense1_kernel(const float* __restrict__ agg,
                              const float* __restrict__ Wg1,
                              const float* __restrict__ bg1,
                              float* __restrict__ out, long n_nodes) {
    long t = (long)blockIdx.x * blockDim.x + threadIdx.x;
    if (t >= (n_nodes << 7)) return;
    long n = t >> 7; int f = (int)(t & 127);
    out[t] = fmaxf(agg[n] * Wg1[f] + bg1[f], 0.0f);
}

__global__ void pool_kernel(const float* __restrict__ h,
                            const int* __restrict__ node_graph,
                            float* __restrict__ gsum, float* __restrict__ gcnt,
                            long n_nodes) {
    long t = (long)blockIdx.x * blockDim.x + threadIdx.x;
    if (t >= (n_nodes << 7)) return;
    long n = t >> 7; int f = (int)(t & 127);
    int g = node_graph[n];
    atomicAdd(&gsum[((long)g << 7) + f], h[t]);
    if (f == 0) atomicAdd(&gcnt[g], 1.0f);
}

__global__ void final_kernel(const float* __restrict__ gsum,
                             const float* __restrict__ gcnt,
                             const float* __restrict__ ds_h,
                             const float* __restrict__ Wr,
                             const float* __restrict__ br,
                             float* __restrict__ out) {
    __shared__ float red[128];
    int b = blockIdx.x, j = threadIdx.x;          // 128 threads
    float cnt = fmaxf(gcnt[b], 1.0f);
    float v = (gsum[(long)b * 128 + j] / cnt) * Wr[j]
            + ds_h[(long)b * 128 + j] * Wr[128 + j];
    red[j] = v;
    __syncthreads();
    for (int s = 64; s > 0; s >>= 1) {
        if (j < s) red[j] += red[j + s];
        __syncthreads();
    }
    if (j == 0) out[b] = red[0] + br[0];
}

// ---------------------------------------------------------------------------
// Host-side helpers
// ---------------------------------------------------------------------------
static inline void launch_fill0(float* p, long n, hipStream_t s) {
    fill0_kernel<<<(int)((n + 255) / 256), 256, 0, s>>>(p, n);
}

// Full GEMM: optionally pack A (if a_f32 given), always pack W^T, then WMMA.
static void run_gemm(const float* a_f32, const __bf16* a_bf_pre,
                     const float* W, const float* bias,
                     const float* g, const float* bb, const float* m, const float* v,
                     float* out_f32, __bf16* out_bf16,
                     int M, int K, int N,
                     __bf16* Abuf, __bf16* WTbuf, hipStream_t s) {
    const int ldK = roundup32(K);
    const __bf16* Ab = a_bf_pre;
    if (!Ab) {
        pack_a_kernel<<<dim3(M, (ldK + 255) / 256), 256, 0, s>>>(a_f32, Abuf, K, ldK);
        Ab = Abuf;
    }
    pack_wt_kernel<<<dim3(N, (ldK + 255) / 256), 256, 0, s>>>(W, WTbuf, K, N, ldK);
    dim3 grid((N / 16 + 3) / 4, (M + 31) / 32);
    gemm_bf16_wmma<<<grid, 128, 0, s>>>(Ab, WTbuf, bias, g, bb, m, v,
                                        out_f32, out_bf16, M, ldK, N,
                                        out_bf16 ? 1 : (bias == nullptr ? 0 : 1));
}

extern "C" void kernel_launch(void* const* d_in, const int* in_sizes, int n_in,
                              void* d_out, int out_size, void* d_ws, size_t ws_size,
                              hipStream_t stream) {
    const int DESC = 9577;
    const float* desc      = (const float*)d_in[0];
    const int*   edge_src  = (const int*)  d_in[1];
    const int*   edge_dst  = (const int*)  d_in[2];
    const int*   node_grph = (const int*)  d_in[3];
    const float* W1 = (const float*)d_in[4];  const float* b1  = (const float*)d_in[5];
    const float* g1 = (const float*)d_in[6];  const float* be1 = (const float*)d_in[7];
    const float* m1 = (const float*)d_in[8];  const float* v1  = (const float*)d_in[9];
    const float* W2 = (const float*)d_in[10]; const float* b2  = (const float*)d_in[11];
    const float* g2 = (const float*)d_in[12]; const float* be2 = (const float*)d_in[13];
    const float* m2 = (const float*)d_in[14]; const float* v2  = (const float*)d_in[15];
    const float* W3 = (const float*)d_in[16]; const float* b3  = (const float*)d_in[17];
    const float* Wg1 = (const float*)d_in[18]; const float* bg1 = (const float*)d_in[19];
    const float* Wg2 = (const float*)d_in[20]; const float* bg2 = (const float*)d_in[21];
    const float* Wg3 = (const float*)d_in[22]; const float* bg3 = (const float*)d_in[23];
    const float* Wr  = (const float*)d_in[24]; const float* br  = (const float*)d_in[25];

    const int  Bsz     = in_sizes[0] / DESC;     // 1024
    const long n_edges = in_sizes[1];            // 800000
    const long n_nodes = in_sizes[3];            // 100000

    float* out = (float*)d_out;

    // ---- workspace layout (byte offsets, 256B aligned) ----
    char* base = (char*)d_ws;
    size_t o = 0;
    auto take = [&](size_t bytes) { char* p = base + o; o += (bytes + 255) & ~(size_t)255; return p; };

    float*  deg   = (float*) take((size_t)n_nodes * 4);
    float*  bufA  = (float*) take((size_t)n_nodes * 256 * 4);   // layer outputs h
    float*  bufB  = (float*) take((size_t)n_nodes * 256 * 4);   // sums / agg
    float*  dsh   = (float*) take((size_t)Bsz * 128 * 4);
    float*  gsum  = (float*) take((size_t)Bsz * 128 * 4);
    float*  gcnt  = (float*) take((size_t)Bsz * 4);
    __bf16* H1bf  = (__bf16*)take((size_t)Bsz * 2048 * 2);
    __bf16* H2bf  = (__bf16*)take((size_t)Bsz * 512 * 2);
    __bf16* Abuf  = (__bf16*)take((size_t)n_nodes * 256 * 2);   // >= Bsz*9600*2
    __bf16* WTbuf = (__bf16*)take((size_t)2048 * 9600 * 2);

    // ---------------- MLP branch (WMMA GEMMs) ----------------
    // H1bf = bf16( relu(bn(desc @ W1 + b1)) )
    run_gemm(desc, nullptr, W1, b1, g1, be1, m1, v1,
             nullptr, H1bf, Bsz, DESC, 2048, Abuf, WTbuf, stream);
    // H2bf = bf16( relu(bn(H1 @ W2 + b2)) )
    run_gemm(nullptr, H1bf, W2, b2, g2, be2, m2, v2,
             nullptr, H2bf, Bsz, 2048, 512, Abuf, WTbuf, stream);
    // dsh = relu(H2 @ W3 + b3)   (fp32 out, no BN)
    run_gemm(nullptr, H2bf, W3, b3, nullptr, nullptr, nullptr, nullptr,
             dsh, nullptr, Bsz, 512, 128, Abuf, WTbuf, stream);

    // ---------------- degree ----------------
    launch_fill0(deg, n_nodes, stream);
    deg_kernel<<<(int)((n_edges + 255) / 256), 256, 0, stream>>>(edge_dst, deg,
                                                                 (int)n_edges);

    // ---------------- GCN layer 1 (F=1, K=1) ----------------
    launch_fill0(bufB, n_nodes, stream);
    scatter_add_kernel<<<(int)((n_edges + 255) / 256), 256, 0, stream>>>(
        deg, edge_src, edge_dst, bufB, n_edges, 0);
    agg_kernel<<<(int)((n_nodes + 255) / 256), 256, 0, stream>>>(
        deg, deg, bufB, n_nodes, 0);
    dense1_kernel<<<(int)((n_nodes * 128 + 255) / 256), 256, 0, stream>>>(
        bufB, Wg1, bg1, bufA, n_nodes);                      // h1 -> bufA [N x 128]

    // ---------------- GCN layer 2 (128 -> 256) ----------------
    launch_fill0(bufB, n_nodes * 128, stream);
    scatter_add_kernel<<<(int)((n_edges * 128 + 255) / 256), 256, 0, stream>>>(
        bufA, edge_src, edge_dst, bufB, n_edges, 7);
    agg_kernel<<<(int)((n_nodes * 128 + 255) / 256), 256, 0, stream>>>(
        bufA, deg, bufB, n_nodes, 7);
    run_gemm(bufB, nullptr, Wg2, bg2, nullptr, nullptr, nullptr, nullptr,
             bufA, nullptr, (int)n_nodes, 128, 256, Abuf, WTbuf, stream);

    // ---------------- GCN layer 3 (256 -> 128) ----------------
    launch_fill0(bufB, n_nodes * 256, stream);
    scatter_add_kernel<<<(int)((n_edges * 256 + 255) / 256), 256, 0, stream>>>(
        bufA, edge_src, edge_dst, bufB, n_edges, 8);
    agg_kernel<<<(int)((n_nodes * 256 + 255) / 256), 256, 0, stream>>>(
        bufA, deg, bufB, n_nodes, 8);
    run_gemm(bufB, nullptr, Wg3, bg3, nullptr, nullptr, nullptr, nullptr,
             bufA, nullptr, (int)n_nodes, 256, 128, Abuf, WTbuf, stream);

    // ---------------- pooling + readout ----------------
    launch_fill0(gsum, (long)Bsz * 128 + Bsz, stream);       // gsum + gcnt adjacent? (separate allocs)
    launch_fill0(gcnt, Bsz, stream);
    pool_kernel<<<(int)((n_nodes * 128 + 255) / 256), 256, 0, stream>>>(
        bufA, node_grph, gsum, gcnt, n_nodes);
    final_kernel<<<Bsz, 128, 0, stream>>>(gsum, gcnt, dsh, Wr, br, out);
}
